// FP8AttentionBlock_32985348833538
// MI455X (gfx1250) — compile-verified, matrix operand order checked by reading, not run on hardware
//
#include <hip/hip_runtime.h>
#include <hip/hip_bf16.h>
#include <math.h>
#include <stdint.h>

// ---------------------------------------------------------------------------
// DiT block (adaLN + RoPE attention + MLP) for gfx1250, wave32, WMMA f16,
// with async global->LDS staging (ASYNCcnt path) for the non-transposed tiles.
// ---------------------------------------------------------------------------

#define DIM     1024
#define HEADS   16
#define HEAD_D  64
#define MLP_D   4096
#define BATCH   2
#define SEQ     2048

typedef __attribute__((ext_vector_type(16))) _Float16 v16h;
typedef __attribute__((ext_vector_type(8)))  _Float16 v8h;
typedef __attribute__((ext_vector_type(8)))  float    v8f;

// Async copy of 16 bytes global -> LDS, tracked by ASYNCcnt (no VGPR data).
__device__ inline void async_copy_b128(const void* gptr, void* lptr) {
  uint32_t lds_off = (uint32_t)(uintptr_t)lptr;   // low 32 bits = LDS offset
  uint64_t gaddr   = (uint64_t)(uintptr_t)gptr;
  asm volatile("global_load_async_to_lds_b128 %0, %1, off"
               :: "v"(lds_off), "v"(gaddr) : "memory");
}
__device__ inline void async_wait0() {
  asm volatile("s_wait_asynccnt 0x0" ::: "memory");
}

// Load one 16-half A/B fragment row from LDS following the CDNA5 16-bit
// operand layout (ISA 7.12.2): lanes 0-15 take k[0..7] and k[16..23],
// lanes 16-31 take k[8..15] and k[24..31]. rowPtr must be 16B aligned.
__device__ inline v16h load_frag_row(const _Float16* rowPtr, int lane) {
  const v8h* p = (const v8h*)rowPtr;
  int hi = (lane < 16) ? 0 : 1;
  union { v16h v; v8h h[2]; } u;
  u.h[0] = p[hi];
  u.h[1] = p[hi + 2];
  return u.v;
}

__device__ inline v8f wmma_f16(v16h a, v16h b, v8f c) {
  return __builtin_amdgcn_wmma_f32_16x16x32_f16(false, a, false, b,
                                                (short)0, c, false, false);
}

__device__ inline float gelu_exact(float x) {
  return 0.5f * x * (1.0f + erff(x * 0.70710678118654752f));
}

// ---------------------------------------------------------------------------
// fp32 -> f16 weight conversion
// ---------------------------------------------------------------------------
__global__ void cvt_f16_kernel(const float* __restrict__ src,
                               _Float16* __restrict__ dst, int n) {
  int i = blockIdx.x * blockDim.x + threadIdx.x;
  if (i < n) dst[i] = (_Float16)src[i];
}

// ---------------------------------------------------------------------------
// mod = silu(t_emb) @ ada_W + ada_b    (B x 6C)
// ---------------------------------------------------------------------------
__global__ void __launch_bounds__(256)
mod_kernel(const float* __restrict__ t_emb, const float* __restrict__ adaW,
           const float* __restrict__ adaB, float* __restrict__ mod) {
  int b = blockIdx.y;
  int j = blockIdx.x * 256 + threadIdx.x;         // 0..6*DIM-1
  __shared__ float silu_s[DIM];
  for (int i = threadIdx.x; i < DIM; i += 256) {
    float t = t_emb[b * DIM + i];
    silu_s[i] = t / (1.0f + __expf(-t));
  }
  __syncthreads();
  float acc = adaB[j];
  for (int i = 0; i < DIM; ++i)
    acc += silu_s[i] * adaW[(size_t)i * (6 * DIM) + j];
  mod[(size_t)b * (6 * DIM) + j] = acc;
}

// ---------------------------------------------------------------------------
// Fused LayerNorm + adaLN modulation -> f16
// ---------------------------------------------------------------------------
__global__ void __launch_bounds__(256)
ln_mod_kernel(const float* __restrict__ x, const float* __restrict__ g,
              const float* __restrict__ beta, const float* __restrict__ mod,
              int scaleOff, int shiftOff, _Float16* __restrict__ out) {
  int row = blockIdx.x;                 // 0 .. B*SEQ-1
  int b = row / SEQ;
  const float* xr = x + (size_t)row * DIM;
  __shared__ float red[256];
  float s = 0.f;
  for (int i = threadIdx.x; i < DIM; i += 256) s += xr[i];
  red[threadIdx.x] = s; __syncthreads();
  for (int st = 128; st > 0; st >>= 1) {
    if (threadIdx.x < st) red[threadIdx.x] += red[threadIdx.x + st];
    __syncthreads();
  }
  float mu = red[0] * (1.0f / DIM);
  __syncthreads();
  float v = 0.f;
  for (int i = threadIdx.x; i < DIM; i += 256) {
    float d = xr[i] - mu; v += d * d;
  }
  red[threadIdx.x] = v; __syncthreads();
  for (int st = 128; st > 0; st >>= 1) {
    if (threadIdx.x < st) red[threadIdx.x] += red[threadIdx.x + st];
    __syncthreads();
  }
  float rsq = rsqrtf(red[0] * (1.0f / DIM) + 1e-5f);
  const float* mrow = mod + (size_t)b * (6 * DIM);
  for (int i = threadIdx.x; i < DIM; i += 256) {
    float xn = (xr[i] - mu) * rsq * g[i] + beta[i];
    out[(size_t)row * DIM + i] =
        (_Float16)(xn * (1.0f + mrow[scaleOff + i]) + mrow[shiftOff + i]);
  }
}

// ---------------------------------------------------------------------------
// Tiled WMMA GEMM:  D = epilogue(A[MxK] @ W[KxN])
//   EP==0: store f16   EP==1: GELU, store f16
//   EP==2: outF = res + gate[b][n] * acc   (fp32)
// Block: 256 threads = 8 waves; tile 128x128, k-step 32.
// A tile staged with async global->LDS (row-major, no transpose needed);
// W tile transposed into LDS via VALU stores.
// ---------------------------------------------------------------------------
template <int EP>
__global__ void __launch_bounds__(256)
gemm_wmma_kernel(const _Float16* __restrict__ A, const _Float16* __restrict__ W,
                 int M, int K, int Nc,
                 _Float16* __restrict__ outH, float* __restrict__ outF,
                 const float* __restrict__ res, const float* __restrict__ gate,
                 int gStride) {
  constexpr int BM = 128, BN = 128, BK = 32, LDA = BK + 16;   // 48 halfs, 96B
  __shared__ __align__(16) _Float16 As[BM * LDA];
  __shared__ __align__(16) _Float16 Bs[BN * LDA];             // Bs[n][k]
  const int tid = threadIdx.x;
  const int lane = tid & 31, wid = tid >> 5;
  const int wm = wid >> 2, wn = wid & 3;
  const int mBase = blockIdx.y * BM, nBase = blockIdx.x * BN;

  v8f c[4][2] = {};

  for (int k0 = 0; k0 < K; k0 += BK) {
    // Stage A tile asynchronously: 128x32 halfs (512 chunks of 16B).
    #pragma unroll
    for (int i = 0; i < 2; ++i) {
      int chunk = tid + i * 256;
      int r = chunk >> 2, kc = chunk & 3;
      async_copy_b128(A + (size_t)(mBase + r) * K + k0 + kc * 8,
                      &As[r * LDA + kc * 8]);
    }
    // Stage W tile transposed into Bs[n][k].
    #pragma unroll
    for (int i = 0; i < 2; ++i) {
      int chunk = tid + i * 256;
      int kk = chunk >> 4, nc = chunk & 15;
      const _Float16* wp = W + (size_t)(k0 + kk) * Nc + nBase + nc * 8;
      v8h w = *(const v8h*)wp;
      #pragma unroll
      for (int j = 0; j < 8; ++j) Bs[(nc * 8 + j) * LDA + kk] = w[j];
    }
    if (k0 + BK < K)  // warm L2 for the next weight panel
      __builtin_prefetch(W + (size_t)(k0 + BK) * Nc + nBase + (tid & 15) * 8, 0, 0);
    async_wait0();          // A tile landed in LDS
    __syncthreads();

    v16h a[4], b[2];
    #pragma unroll
    for (int mt = 0; mt < 4; ++mt)
      a[mt] = load_frag_row(&As[(wm * 64 + mt * 16 + (lane & 15)) * LDA], lane);
    #pragma unroll
    for (int nt = 0; nt < 2; ++nt)
      b[nt] = load_frag_row(&Bs[(wn * 32 + nt * 16 + (lane & 15)) * LDA], lane);

    #pragma unroll
    for (int mt = 0; mt < 4; ++mt)
      #pragma unroll
      for (int nt = 0; nt < 2; ++nt)
        c[mt][nt] = wmma_f16(a[mt], b[nt], c[mt][nt]);
    __syncthreads();
  }

  // Epilogue. C layout: lane<16 -> n=lane, rows 0..7; lane>=16 -> n=lane-16, rows 8..15.
  const int nLocal = lane & 15;
  const int rOff   = (lane >= 16) ? 8 : 0;
  #pragma unroll
  for (int mt = 0; mt < 4; ++mt) {
    #pragma unroll
    for (int nt = 0; nt < 2; ++nt) {
      int n = nBase + wn * 32 + nt * 16 + nLocal;
      #pragma unroll
      for (int i = 0; i < 8; ++i) {
        int m = mBase + wm * 64 + mt * 16 + rOff + i;
        float v = c[mt][nt][i];
        if (EP == 1) v = gelu_exact(v);
        if (EP <= 1) {
          outH[(size_t)m * Nc + n] = (_Float16)v;
        } else {
          int bb = m / SEQ;
          outF[(size_t)m * Nc + n] =
              res[(size_t)m * Nc + n] + gate[bb * gStride + n] * v;
        }
      }
    }
  }
}

// ---------------------------------------------------------------------------
// RoPE (reference convention: rotate_half of interleaved pairs, concat freqs)
// ---------------------------------------------------------------------------
__global__ void __launch_bounds__(256)
rope_kernel(_Float16* __restrict__ x) {
  int idx = blockIdx.x * blockDim.x + threadIdx.x;
  if (idx >= BATCH * SEQ * HEADS) return;
  int h = idx % HEADS;
  int n = (idx / HEADS) % SEQ;
  int b = idx / (HEADS * SEQ);
  _Float16* p = x + ((size_t)(b * SEQ + n)) * DIM + h * HEAD_D;
  float buf[HEAD_D];
  #pragma unroll
  for (int d = 0; d < HEAD_D; ++d) buf[d] = (float)p[d];
  #pragma unroll
  for (int j = 0; j < HEAD_D / 2; ++j) {
    float inv = __powf(10000.0f, -(float)j / (HEAD_D / 2));
    float ang = (float)n * inv;
    float c = __cosf(ang), s = __sinf(ang);
    float o1 = buf[j] * c - buf[2 * j + 1] * s;          // out[j]
    float o2 = buf[32 + j] * c + buf[2 * j] * s;         // out[32+j]
    p[j]      = (_Float16)o1;
    p[32 + j] = (_Float16)o2;
  }
}

// ---------------------------------------------------------------------------
// Flash attention, one (b, h, 64-query block) per workgroup, 4 waves.
// K staged async global->LDS; V transposed into Vt[d][n]; QK^T and P*V via
// v_wmma_f32_16x16x32_f16; online softmax with 16-lane shfl_xor reductions.
// ---------------------------------------------------------------------------
__global__ void __launch_bounds__(128)
flash_attn_kernel(const _Float16* __restrict__ q, const _Float16* __restrict__ k,
                  const _Float16* __restrict__ v, _Float16* __restrict__ o) {
  constexpr int LKV = HEAD_D + 16;                 // 80 halfs = 160B (16B mult)
  __shared__ __align__(16) _Float16 Ks[64 * LKV];  // Ks[n][d]
  __shared__ __align__(16) _Float16 Vt[64 * LKV];  // Vt[d][n]
  __shared__ __align__(16) _Float16 Ps[4][16 * LKV];

  const int tid = threadIdx.x, lane = tid & 31, wid = tid >> 5;
  const int b = blockIdx.z, h = blockIdx.y, q0 = blockIdx.x * 64;
  const size_t headOff = (size_t)h * HEAD_D;
  const float scale = 0.125f;                      // 1/sqrt(64)

  // Q fragments for this wave's 16 rows (A layout), d-chunks 0..31 / 32..63.
  v16h aq[2];
  {
    int mrow = q0 + wid * 16 + (lane & 15);
    const v8h* p = (const v8h*)(q + ((size_t)(b * SEQ + mrow)) * DIM + headOff);
    int hi = (lane < 16) ? 0 : 1;
    union { v16h vv; v8h hh[2]; } u0, u1;
    u0.hh[0] = p[hi];     u0.hh[1] = p[hi + 2];
    u1.hh[0] = p[4 + hi]; u1.hh[1] = p[4 + hi + 2];
    aq[0] = u0.vv; aq[1] = u1.vv;
  }

  float mrun[8], lrun[8];
  v8f ofr[4] = {};
  #pragma unroll
  for (int i = 0; i < 8; ++i) { mrun[i] = -1e30f; lrun[i] = 0.f; }

  for (int kv0 = 0; kv0 < SEQ; kv0 += 64) {
    // Stage K (async, no transpose) and V (transposed): 64x64 halfs each.
    #pragma unroll
    for (int i = 0; i < 4; ++i) {
      int chunk = tid + i * 128;
      int n = chunk >> 3, dc = chunk & 7;
      size_t gi = ((size_t)(b * SEQ + kv0 + n)) * DIM + headOff + dc * 8;
      async_copy_b128(k + gi, &Ks[n * LKV + dc * 8]);
      v8h vv = *(const v8h*)(v + gi);
      #pragma unroll
      for (int j = 0; j < 8; ++j) Vt[(dc * 8 + j) * LKV + n] = vv[j];
    }
    async_wait0();
    __syncthreads();

    // S = scale * Q K^T  (4 n-tiles of 16 keys, 2 k-chunks over d)
    v8f s[4];
    #pragma unroll
    for (int nt = 0; nt < 4; ++nt) {
      const _Float16* krow = &Ks[(nt * 16 + (lane & 15)) * LKV];
      v8f acc = {};
      acc = wmma_f16(aq[0], load_frag_row(krow, lane), acc);
      acc = wmma_f16(aq[1], load_frag_row(krow + 32, lane), acc);
      s[nt] = acc;
    }

    // Online softmax: per-row max/sum across the 16 lanes holding a row.
    float mnew[8];
    #pragma unroll
    for (int i = 0; i < 8; ++i) {
      float mx = -1e30f;
      #pragma unroll
      for (int nt = 0; nt < 4; ++nt) { s[nt][i] *= scale; mx = fmaxf(mx, s[nt][i]); }
      #pragma unroll
      for (int off = 1; off < 16; off <<= 1) mx = fmaxf(mx, __shfl_xor(mx, off, 32));
      mnew[i] = fmaxf(mrun[i], mx);
    }
    #pragma unroll
    for (int i = 0; i < 8; ++i) {
      float alpha = __expf(mrun[i] - mnew[i]);
      float rsum = 0.f;
      #pragma unroll
      for (int nt = 0; nt < 4; ++nt) {
        float pij = __expf(s[nt][i] - mnew[i]);
        s[nt][i] = pij; rsum += pij;
      }
      #pragma unroll
      for (int off = 1; off < 16; off <<= 1) rsum += __shfl_xor(rsum, off, 32);
      lrun[i] = lrun[i] * alpha + rsum;
      mrun[i] = mnew[i];
      #pragma unroll
      for (int dt = 0; dt < 4; ++dt) ofr[dt][i] *= alpha;
    }

    // Route P through wave-private LDS (C layout -> A layout).
    {
      int n = lane & 15, rb = (lane >= 16) ? 8 : 0;
      #pragma unroll
      for (int nt = 0; nt < 4; ++nt)
        #pragma unroll
        for (int i = 0; i < 8; ++i)
          Ps[wid][(rb + i) * LKV + nt * 16 + n] = (_Float16)s[nt][i];
    }
    asm volatile("s_wait_dscnt 0x0" ::: "memory");  // wave-local LDS visibility

    // O += P V   (A = P over n-chunks, B = Vt rows = V columns)
    #pragma unroll
    for (int nc = 0; nc < 2; ++nc) {
      v16h ap = load_frag_row(&Ps[wid][(lane & 15) * LKV + nc * 32], lane);
      #pragma unroll
      for (int dt = 0; dt < 4; ++dt) {
        v16h bv = load_frag_row(&Vt[(dt * 16 + (lane & 15)) * LKV + nc * 32], lane);
        ofr[dt] = wmma_f16(ap, bv, ofr[dt]);
      }
    }
    __syncthreads();   // protect Ks/Vt before next staging
  }

  // Normalize and store (B, n, h*64 + d) as f16.
  {
    int n = lane & 15, rb = (lane >= 16) ? 8 : 0;
    #pragma unroll
    for (int dt = 0; dt < 4; ++dt)
      #pragma unroll
      for (int i = 0; i < 8; ++i) {
        int row = q0 + wid * 16 + rb + i;
        float val = ofr[dt][i] / lrun[i];
        o[((size_t)(b * SEQ + row)) * DIM + headOff + dt * 16 + n] = (_Float16)val;
      }
  }
}

// ---------------------------------------------------------------------------
// Launch
// ---------------------------------------------------------------------------
extern "C" void kernel_launch(void* const* d_in, const int* in_sizes, int n_in,
                              void* d_out, int out_size, void* d_ws, size_t ws_size,
                              hipStream_t stream) {
  const float* x     = (const float*)d_in[0];
  const float* t_emb = (const float*)d_in[1];
  const float* n1g   = (const float*)d_in[2];
  const float* n1b   = (const float*)d_in[3];
  const float* Wq    = (const float*)d_in[4];
  const float* Wk    = (const float*)d_in[5];
  const float* Wv    = (const float*)d_in[6];
  const float* Wo    = (const float*)d_in[7];
  const float* n2g   = (const float*)d_in[8];
  const float* n2b   = (const float*)d_in[9];
  const float* W1    = (const float*)d_in[10];
  const float* W2    = (const float*)d_in[11];
  const float* adaW  = (const float*)d_in[12];
  const float* adaB  = (const float*)d_in[13];
  float* out = (float*)d_out;

  const int M = BATCH * SEQ;                 // 4096 rows
  char* ws = (char*)d_ws;
  size_t off = 0;
  auto alloc = [&](size_t bytes) -> void* {
    void* p = ws + off;
    off = (off + bytes + 255) & ~(size_t)255;
    return p;
  };
  float*    mod   = (float*)   alloc(sizeof(float) * BATCH * 6 * DIM);
  _Float16* WqH   = (_Float16*)alloc(sizeof(_Float16) * DIM * DIM);
  _Float16* WkH   = (_Float16*)alloc(sizeof(_Float16) * DIM * DIM);
  _Float16* WvH   = (_Float16*)alloc(sizeof(_Float16) * DIM * DIM);
  _Float16* WoH   = (_Float16*)alloc(sizeof(_Float16) * DIM * DIM);
  _Float16* W1H   = (_Float16*)alloc(sizeof(_Float16) * DIM * MLP_D);
  _Float16* W2H   = (_Float16*)alloc(sizeof(_Float16) * MLP_D * DIM);
  _Float16* xnH   = (_Float16*)alloc(sizeof(_Float16) * M * DIM);
  _Float16* qH    = (_Float16*)alloc(sizeof(_Float16) * M * DIM);
  _Float16* kH    = (_Float16*)alloc(sizeof(_Float16) * M * DIM);
  _Float16* vH    = (_Float16*)alloc(sizeof(_Float16) * M * DIM);
  _Float16* attnH = (_Float16*)alloc(sizeof(_Float16) * M * DIM);
  float*    x1    = (float*)   alloc(sizeof(float) * M * DIM);
  _Float16* xn2H  = (_Float16*)alloc(sizeof(_Float16) * M * DIM);
  _Float16* h1H   = (_Float16*)alloc(sizeof(_Float16) * M * MLP_D);

  auto cvt = [&](const float* s, _Float16* d, int n) {
    cvt_f16_kernel<<<(n + 255) / 256, 256, 0, stream>>>(s, d, n);
  };
  cvt(Wq, WqH, DIM * DIM);  cvt(Wk, WkH, DIM * DIM);
  cvt(Wv, WvH, DIM * DIM);  cvt(Wo, WoH, DIM * DIM);
  cvt(W1, W1H, DIM * MLP_D); cvt(W2, W2H, MLP_D * DIM);

  // adaLN modulation: shift_msa|scale_msa|gate_msa|shift_mlp|scale_mlp|gate_mlp
  mod_kernel<<<dim3(6 * DIM / 256, BATCH), 256, 0, stream>>>(t_emb, adaW, adaB, mod);

  // xn = LN1(x)*(1+scale_msa)+shift_msa
  ln_mod_kernel<<<M, 256, 0, stream>>>(x, n1g, n1b, mod, /*scale*/ DIM, /*shift*/ 0, xnH);

  // Q/K/V projections
  dim3 gProj(DIM / 128, M / 128);
  gemm_wmma_kernel<0><<<gProj, 256, 0, stream>>>(xnH, WqH, M, DIM, DIM, qH, nullptr, nullptr, nullptr, 0);
  gemm_wmma_kernel<0><<<gProj, 256, 0, stream>>>(xnH, WkH, M, DIM, DIM, kH, nullptr, nullptr, nullptr, 0);
  gemm_wmma_kernel<0><<<gProj, 256, 0, stream>>>(xnH, WvH, M, DIM, DIM, vH, nullptr, nullptr, nullptr, 0);

  // RoPE on q and k
  int rthreads = BATCH * SEQ * HEADS;
  rope_kernel<<<(rthreads + 255) / 256, 256, 0, stream>>>(qH);
  rope_kernel<<<(rthreads + 255) / 256, 256, 0, stream>>>(kH);

  // Flash attention
  flash_attn_kernel<<<dim3(SEQ / 64, HEADS, BATCH), 128, 0, stream>>>(qH, kH, vH, attnH);

  // x1 = x + gate_msa * (attn @ Wo)
  gemm_wmma_kernel<2><<<gProj, 256, 0, stream>>>(attnH, WoH, M, DIM, DIM,
                                                 nullptr, x1, x, mod + 2 * DIM, 6 * DIM);

  // xn2 = LN2(x1)*(1+scale_mlp)+shift_mlp
  ln_mod_kernel<<<M, 256, 0, stream>>>(x1, n2g, n2b, mod, /*scale*/ 4 * DIM, /*shift*/ 3 * DIM, xn2H);

  // h = gelu(xn2 @ W1)
  gemm_wmma_kernel<1><<<dim3(MLP_D / 128, M / 128), 256, 0, stream>>>(
      xn2H, W1H, M, DIM, MLP_D, h1H, nullptr, nullptr, nullptr, 0);

  // out = x1 + gate_mlp * (h @ W2)
  gemm_wmma_kernel<2><<<dim3(DIM / 128, M / 128), 256, 0, stream>>>(
      h1H, W2H, M, MLP_D, DIM, nullptr, out, x1, mod + 5 * DIM, 6 * DIM);

  (void)in_sizes; (void)n_in; (void)out_size; (void)ws_size;
}
